// GCN_53592601920045
// MI455X (gfx1250) — matrix operand order repaired and checked by reading, not compiled, not run on hardware
//
#include <hip/hip_runtime.h>

// ---------------------------------------------------------------------------
// 2-layer GCN forward for MI455X (gfx1250, wave32).
//   h1   = X @ W1                          (WMMA bf16, f32 accumulate)
//   agg1 = scatter_add(norm * h1[src] -> dst) + selfloop + b1 ; relu
//   h2   = agg1 @ W2                       (WMMA bf16)
//   out  = scatter_add(norm * h2[src] -> dst) + selfloop + b2
// norm = dinv[src]*dinv[dst], dinv = rsqrt(1 + in_degree).
// Whole working set (~100 MB incl. inputs) fits in the 192 MB L2, so the
// scatter phases run at L2 atomic throughput (native global_atomic_add_f32);
// GEMMs (2.4 GFLOP) are trivial on the WMMA units.
// ---------------------------------------------------------------------------

typedef __attribute__((ext_vector_type(16))) __bf16 v16bf;
typedef __attribute__((ext_vector_type(8)))  float  v8f;

#define D_IN  128
#define D_HID 128
#define D_OUT 64

static __device__ __forceinline__ void atomAdd(float* p, float v) {
    // HIP's unsafeAtomicAdd maps to the hardware f32 fadd
    // (global_atomic_add_f32) unconditionally — no CAS fallback.
    unsafeAtomicAdd(p, v);
}

// ---------------------------------------------------------------- fill -----
__global__ __launch_bounds__(256) void fill_kernel(float* __restrict__ p,
                                                   long n, float v) {
    long i = (long)blockIdx.x * blockDim.x + threadIdx.x;
    long stride = (long)gridDim.x * blockDim.x;
    for (; i < n; i += stride) p[i] = v;
}

// -------------------------------------------------------------- degree -----
__global__ __launch_bounds__(256) void degree_kernel(const int* __restrict__ dst,
                                                     int E, float* __restrict__ deg) {
    int e = blockIdx.x * blockDim.x + threadIdx.x;
    if (e < E) atomAdd(&deg[dst[e]], 1.0f);   // counts are exact in f32
}

__global__ __launch_bounds__(256) void rsqrt_kernel(const float* __restrict__ deg,
                                                    float* __restrict__ dinv, int n) {
    int i = blockIdx.x * blockDim.x + threadIdx.x;
    if (i < n) dinv[i] = rsqrtf(deg[i]);      // deg >= 1 (self loop)
}

// ---------------------------------------------------------- WMMA GEMM ------
// C[nrows x NCOLS] = A[nrows x 128] * B[128 x NCOLS], all f32 row-major.
// One wave computes one 16x16 tile; K=128 in 4 steps of 32 via bf16 WMMA.
template <int NCOLS>
__global__ __launch_bounds__(256) void gemm_wmma(const float* __restrict__ A,
                                                 const float* __restrict__ B,
                                                 float* __restrict__ C,
                                                 int nrows) {
    const int lane = threadIdx.x & 31;
    const int wave = threadIdx.x >> 5;
    const int ntiles = NCOLS / 16;
    const long mtiles = nrows / 16;                  // nrows multiple of 16
    long gw = (long)blockIdx.x * (blockDim.x >> 5) + wave;
    if (gw >= mtiles * ntiles) return;
    const int mt = (int)(gw / ntiles);
    const int nt = (int)(gw % ntiles);
    const int m0 = mt * 16, n0 = nt * 16;
    const int half = lane >> 4;                      // 0: lanes 0-15, 1: 16-31
    const int mrow = m0 + (lane & 15);

    v8f acc = {};
    const float* arow = A + (size_t)mrow * 128;

#pragma unroll
    for (int ks = 0; ks < 4; ++ks) {
        const int k0 = ks * 32;

        // A tile (16x32 bf16): lane=M; VGPR v: pairs K = {2v,2v+1}(+8*half),
        // v>=4 offset by 16 (ISA 7.12.2, 16-bit A 16x32 layout).
        v16bf a;
#pragma unroll
        for (int v = 0; v < 8; ++v) {
            const int kbase = (v < 4 ? 2 * v : 16 + 2 * (v - 4)) + half * 8;
            float2 p = *(const float2*)(arow + k0 + kbase);
            a[2 * v]     = (__bf16)p.x;
            a[2 * v + 1] = (__bf16)p.y;
        }

        // B tile (32x16 bf16): lane%16 = N; element s -> K in same order as A.
        const float* bcol = B + (size_t)k0 * NCOLS + n0 + (lane & 15);
        v16bf b;
#pragma unroll
        for (int s = 0; s < 16; ++s) {
            const int kk = (s < 8 ? s : s + 8) + half * 8;
            b[s] = (__bf16)bcol[(size_t)kk * NCOLS];
        }

        acc = __builtin_amdgcn_wmma_f32_16x16x32_bf16(
            false, a, false, b, (short)0, acc, false, false);
    }

    // C/D layout: lane = N (mod 16), VGPR r -> M = r + 8*half.
#pragma unroll
    for (int r = 0; r < 8; ++r) {
        const int row = m0 + r + 8 * half;
        C[(size_t)row * NCOLS + n0 + (lane & 15)] = acc[r];
    }
}

// ------------------------------------------------------------- scatter -----
// One wave per edge: lane owns D/32 contiguous columns; vector gather of
// h[src], native f32 atomic adds into agg[dst].
template <int D>
__global__ __launch_bounds__(256) void scatter_kernel(const float* __restrict__ h,
                                                      const int* __restrict__ src,
                                                      const int* __restrict__ dst,
                                                      const float* __restrict__ dinv,
                                                      float* __restrict__ agg, int E) {
    const int lane = threadIdx.x & 31;
    const int wave = threadIdx.x >> 5;
    long e = (long)blockIdx.x * (blockDim.x >> 5) + wave;
    if (e >= E) return;
    const int s = src[e], d = dst[e];
    const float w = dinv[s] * dinv[d];
    constexpr int CPL = D / 32;                       // 4 (128) or 2 (64)
    const float* hp = h + (size_t)s * D + lane * CPL;
    float* ap = agg + (size_t)d * D + lane * CPL;
    if constexpr (CPL == 4) {
        float4 v = *(const float4*)hp;                // global_load_b128
        atomAdd(ap + 0, v.x * w);
        atomAdd(ap + 1, v.y * w);
        atomAdd(ap + 2, v.z * w);
        atomAdd(ap + 3, v.w * w);
    } else {
        float2 v = *(const float2*)hp;                // global_load_b64
        atomAdd(ap + 0, v.x * w);
        atomAdd(ap + 1, v.y * w);
    }
}

// ------------------------------------------------------------ finalize -----
// out[t] = (relu?)( agg[t] + hself[t]*dinv[i]^2 + bias[c] ),  i=t/D, c=t%D.
template <int D, bool RELU>
__global__ __launch_bounds__(256) void finalize_kernel(const float* __restrict__ agg,
                                                       const float* __restrict__ hself,
                                                       const float* __restrict__ dinv,
                                                       const float* __restrict__ bias,
                                                       float* __restrict__ out, long n) {
    long t = (long)blockIdx.x * blockDim.x + threadIdx.x;
    long stride = (long)gridDim.x * blockDim.x;
    for (; t < n; t += stride) {
        const long i = t / D;
        const int  c = (int)(t % D);
        const float di = dinv[i];
        float v = agg[t] + hself[t] * di * di + bias[c];
        if (RELU) v = v > 0.0f ? v : 0.0f;
        out[t] = v;
    }
}

// ---------------------------------------------------------------------------
extern "C" void kernel_launch(void* const* d_in, const int* in_sizes, int n_in,
                              void* d_out, int out_size, void* d_ws, size_t ws_size,
                              hipStream_t stream) {
    const float* x  = (const float*)d_in[0];
    const int*   ei = (const int*)d_in[1];
    const float* W1 = (const float*)d_in[2];
    const float* b1 = (const float*)d_in[3];
    const float* W2 = (const float*)d_in[4];
    const float* b2 = (const float*)d_in[5];
    float* out = (float*)d_out;

    const int N = in_sizes[0] / D_IN;      // 50000 (multiple of 16)
    const int E = in_sizes[1] / 2;         // 800000
    const int* src = ei;
    const int* dst = ei + E;

    // Workspace layout (floats), ~52 MB total:
    //   deg[Na] | dinv[Na] | h1[N*128] | agg1[N*128]
    //   h1 region is recycled after layer-1 finalize: h2 = h1, agg2 = h1+N*64.
    float* ws = (float*)d_ws;
    const size_t Na = ((size_t)N + 63) & ~(size_t)63;
    float* deg  = ws;
    float* dinv = deg + Na;
    float* h1   = dinv + Na;
    float* agg1 = h1 + (size_t)N * D_HID;  // becomes relu(h1-layer output) in place
    float* h2   = h1;                      // N*64, reuse (h1 dead after finalize1)
    float* agg2 = h1 + (size_t)N * D_OUT;  // second half of old h1 region

    const long n1 = (long)N * D_HID;
    const long n2 = (long)N * D_OUT;

    // --- norm ---
    fill_kernel<<<512, 256, 0, stream>>>(deg, N, 1.0f);              // self loops
    fill_kernel<<<4096, 256, 0, stream>>>(agg1, n1, 0.0f);
    degree_kernel<<<(E + 255) / 256, 256, 0, stream>>>(dst, E, deg);
    rsqrt_kernel<<<(N + 255) / 256, 256, 0, stream>>>(deg, dinv, N);

    // --- layer 1 ---
    {
        const int waves = (N / 16) * (D_HID / 16);                   // 25000
        gemm_wmma<D_HID><<<(waves + 7) / 8, 256, 0, stream>>>(x, W1, h1, N);
    }
    scatter_kernel<D_HID><<<(E + 7) / 8, 256, 0, stream>>>(h1, src, dst, dinv, agg1, E);
    finalize_kernel<D_HID, true><<<4096, 256, 0, stream>>>(agg1, h1, dinv, b1, agg1, n1);

    // --- layer 2 ---
    fill_kernel<<<4096, 256, 0, stream>>>(agg2, n2, 0.0f);
    {
        const int waves = (N / 16) * (D_OUT / 16);                   // 12500
        gemm_wmma<D_OUT><<<(waves + 7) / 8, 256, 0, stream>>>(agg1, W2, h2, N);
    }
    scatter_kernel<D_OUT><<<(E + 7) / 8, 256, 0, stream>>>(h2, src, dst, dinv, agg2, E);
    finalize_kernel<D_OUT, false><<<4096, 256, 0, stream>>>(agg2, h2, dinv, b2, out, n2);
}
